// GAT_47802986005030
// MI455X (gfx1250) — matrix operand order repaired and checked by reading, not compiled, not run on hardware
//
#include <hip/hip_runtime.h>
#include <hip/hip_bf16.h>

// ---------------- problem constants (match reference) ----------------
#define N_NODES 50000
#define IN_DIM  256
#define H1      4
#define C1      32
#define HC1     128      // H1*C1
#define OUT_C   40
#define OUT_PAD 48       // padded to 3 WMMA column tiles
#define NEG_SLOPE 0.2f

typedef __attribute__((ext_vector_type(16))) __bf16 v16bf;
typedef __attribute__((ext_vector_type(8)))  float  v8f;

union Frag16 { v16bf v; unsigned int u[8]; };

__device__ __forceinline__ unsigned short f2bf(float f) {
    unsigned int u = __float_as_uint(f);
    unsigned int r = u + 0x7FFFu + ((u >> 16) & 1u);   // round-to-nearest-even
    return (unsigned short)(r >> 16);
}

// order-preserving float <-> uint map (for atomicMax-based segment max)
__device__ __forceinline__ unsigned int encF(float x) {
    unsigned int u = __float_as_uint(x);
    return (u & 0x80000000u) ? ~u : (u | 0x80000000u);
}
__device__ __forceinline__ float decF(unsigned int e) {
    return __uint_as_float((e & 0x80000000u) ? (e & 0x7FFFFFFFu) : ~e);
}

// ============================================================================
// GEMM1: h1[N,128] = x[N,256] @ W1[256,128]   (bf16 WMMA, f32 accum)
// block = 256 threads = 8 waves; block tile 128 rows x 128 cols;
// wave tile 16 rows x 128 cols (8 accumulators); K loop 8 x 32.
// ============================================================================
__global__ __launch_bounds__(256) void gemm1_wmma(
    const float* __restrict__ x, const float* __restrict__ W1,
    float* __restrict__ h1)
{
    __shared__ unsigned short lA[128 * 36];   // [row][k] bf16, stride 36
    __shared__ unsigned short lB[128 * 36];   // [col][k] bf16, stride 36

    const int t    = threadIdx.x;
    const int wave = t >> 5;
    const int lane = t & 31;
    const int half = lane >> 4;
    const int l16  = lane & 15;
    const int rowBlock = blockIdx.x * 128;

    v8f acc[8];
    const v8f vzero = {0.f,0.f,0.f,0.f,0.f,0.f,0.f,0.f};
#pragma unroll
    for (int i = 0; i < 8; ++i) acc[i] = vzero;

    for (int ks = 0; ks < 8; ++ks) {
        const int k0 = ks * 32;
        __syncthreads();
        // ---- stage A tile: 128 rows x 32 k (f32 -> bf16) ----
#pragma unroll
        for (int i = 0; i < 4; ++i) {
            int gi  = t + i * 256;            // 1024 float4 slots
            int row = gi >> 3;
            int kq  = gi & 7;
            float4 v = make_float4(0.f, 0.f, 0.f, 0.f);
            int grow = rowBlock + row;
            if (grow < N_NODES)
                v = *(const float4*)(x + (size_t)grow * IN_DIM + k0 + kq * 4);
            unsigned int p0 = (unsigned int)f2bf(v.x) | ((unsigned int)f2bf(v.y) << 16);
            unsigned int p1 = (unsigned int)f2bf(v.z) | ((unsigned int)f2bf(v.w) << 16);
            unsigned int* d = (unsigned int*)&lA[row * 36 + kq * 4];
            d[0] = p0; d[1] = p1;
        }
        // ---- stage B tile transposed: W1[k0..k0+31, 0..127] -> lB[col][k] ----
        {
            int col  = t & 127;
            int ksub = t >> 7;                // 0..1
#pragma unroll
            for (int kk = 0; kk < 16; ++kk) {
                int k = kk * 2 + ksub;
                lB[col * 36 + k] = f2bf(W1[(size_t)(k0 + k) * HC1 + col]);
            }
        }
        __syncthreads();

        // ---- A fragment (16x32 bf16, ISA layout) ----
        Frag16 aF;
#pragma unroll
        for (int i = 0; i < 8; ++i) {
            int kf = ((i < 4) ? (2 * i) : (16 + 2 * (i - 4))) + 8 * half;
            aF.u[i] = *(const unsigned int*)&lA[(wave * 16 + l16) * 36 + kf];
        }
        // ---- 8 column tiles ----
#pragma unroll
        for (int ct = 0; ct < 8; ++ct) {
            Frag16 bF;
            int col = ct * 16 + l16;
#pragma unroll
            for (int i = 0; i < 8; ++i) {
                int kf = ((i < 4) ? (2 * i) : (16 + 2 * (i - 4))) + 8 * half;
                bF.u[i] = *(const unsigned int*)&lB[col * 36 + kf];
            }
            acc[ct] = __builtin_amdgcn_wmma_f32_16x16x32_bf16(
                false, aF.v, false, bF.v, (short)0, acc[ct], false, false);
        }
    }

    // ---- store C (VGPR r -> row r + 8*half, col = lane&15) ----
#pragma unroll
    for (int ct = 0; ct < 8; ++ct) {
        int col = ct * 16 + l16;
#pragma unroll
        for (int r = 0; r < 8; ++r) {
            int row = rowBlock + wave * 16 + r + 8 * half;
            if (row < N_NODES) h1[(size_t)row * HC1 + col] = acc[ct][r];
        }
    }
}

// ============================================================================
// GEMM2: g[N,40] = h2[N,128] @ W2pad[128,48]  (bf16 WMMA)
// B (12KB bf16) staged to LDS once; K loop 4 x 32; wave tile 16 x 48.
// ============================================================================
__global__ __launch_bounds__(256) void gemm2_wmma(
    const float* __restrict__ h2, const unsigned short* __restrict__ W2bf,
    float* __restrict__ g)
{
    __shared__ unsigned short lA[128 * 36];    // [row][k]
    __shared__ unsigned short lB[OUT_PAD * 132]; // [col][k], full K=128, stride 132

    const int t    = threadIdx.x;
    const int wave = t >> 5;
    const int lane = t & 31;
    const int half = lane >> 4;
    const int l16  = lane & 15;
    const int rowBlock = blockIdx.x * 128;

    for (int i = t; i < 128 * OUT_PAD; i += 256) {
        int k = i / OUT_PAD, col = i % OUT_PAD;
        lB[col * 132 + k] = W2bf[i];
    }

    v8f acc[3];
    const v8f vzero = {0.f,0.f,0.f,0.f,0.f,0.f,0.f,0.f};
#pragma unroll
    for (int i = 0; i < 3; ++i) acc[i] = vzero;

    for (int ks = 0; ks < 4; ++ks) {
        const int k0 = ks * 32;
        __syncthreads();
#pragma unroll
        for (int i = 0; i < 4; ++i) {
            int gi  = t + i * 256;
            int row = gi >> 3;
            int kq  = gi & 7;
            float4 v = make_float4(0.f, 0.f, 0.f, 0.f);
            int grow = rowBlock + row;
            if (grow < N_NODES)
                v = *(const float4*)(h2 + (size_t)grow * HC1 + k0 + kq * 4);
            unsigned int p0 = (unsigned int)f2bf(v.x) | ((unsigned int)f2bf(v.y) << 16);
            unsigned int p1 = (unsigned int)f2bf(v.z) | ((unsigned int)f2bf(v.w) << 16);
            unsigned int* d = (unsigned int*)&lA[row * 36 + kq * 4];
            d[0] = p0; d[1] = p1;
        }
        __syncthreads();

        Frag16 aF;
#pragma unroll
        for (int i = 0; i < 8; ++i) {
            int kf = ((i < 4) ? (2 * i) : (16 + 2 * (i - 4))) + 8 * half;
            aF.u[i] = *(const unsigned int*)&lA[(wave * 16 + l16) * 36 + kf];
        }
#pragma unroll
        for (int ct = 0; ct < 3; ++ct) {
            Frag16 bF;
            int col = ct * 16 + l16;
#pragma unroll
            for (int i = 0; i < 8; ++i) {
                int kf = k0 + ((i < 4) ? (2 * i) : (16 + 2 * (i - 4))) + 8 * half;
                bF.u[i] = *(const unsigned int*)&lB[col * 132 + kf];
            }
            acc[ct] = __builtin_amdgcn_wmma_f32_16x16x32_bf16(
                false, aF.v, false, bF.v, (short)0, acc[ct], false, false);
        }
    }

#pragma unroll
    for (int ct = 0; ct < 3; ++ct) {
        int col = ct * 16 + l16;
#pragma unroll
        for (int r = 0; r < 8; ++r) {
            int row = rowBlock + wave * 16 + r + 8 * half;
            if (row < N_NODES && col < OUT_C) g[(size_t)row * OUT_C + col] = acc[ct][r];
        }
    }
}

// ============================================================================
// glue kernels
// ============================================================================
__global__ void k_fill_u32(unsigned int* p, int n, unsigned int val) {
    int i = blockIdx.x * blockDim.x + threadIdx.x;
    if (i < n) p[i] = val;
}

// attention-logit coefficients: al[n,h] = sum_c feat[n,h,c]*a[h,c]
__global__ void k_al(const float* __restrict__ feat,
                     const float* __restrict__ aS, const float* __restrict__ aD,
                     float* __restrict__ alS, float* __restrict__ alD,
                     int H, int C) {
    int tid = blockIdx.x * blockDim.x + threadIdx.x;
    if (tid >= N_NODES * H) return;
    int n = tid / H, hd = tid % H;
    const float* row = feat + (size_t)n * (H * C) + hd * C;
    float s = 0.f, d = 0.f;
    for (int c = 0; c < C; ++c) {
        float v = row[c];
        s += v * aS[hd * C + c];
        d += v * aD[hd * C + c];
    }
    alS[tid] = s; alD[tid] = d;
}

// per-edge leaky-relu logit + segment max (encoded-uint atomicMax)
__global__ void k_logit_max(const int* __restrict__ srcA, const int* __restrict__ dstA,
                            const float* __restrict__ alS, const float* __restrict__ alD,
                            float* __restrict__ elog, unsigned int* __restrict__ mEnc,
                            int H, int E, int EN) {
    int tid = blockIdx.x * blockDim.x + threadIdx.x;
    if (tid >= EN * H) return;
    int e = tid / H, h = tid % H;
    int s, d;
    if (e < E) { s = srcA[e]; d = dstA[e]; } else { s = d = e - E; }
    float l = alS[s * H + h] + alD[d * H + h];
    l = (l > 0.f) ? l : NEG_SLOPE * l;
    elog[tid] = l;
    atomicMax(&mEnc[d * H + h], encF(l));
}

// p = exp(logit - max); segment sum
__global__ void k_expsum(const int* __restrict__ srcA, const int* __restrict__ dstA,
                         float* __restrict__ elog, const unsigned int* __restrict__ mEnc,
                         float* __restrict__ ssum, int H, int E, int EN) {
    int tid = blockIdx.x * blockDim.x + threadIdx.x;
    if (tid >= EN * H) return;
    int e = tid / H, h = tid % H;
    int d = (e < E) ? dstA[e] : (e - E);
    float p = __expf(elog[tid] - decF(mEnc[d * H + h]));
    elog[tid] = p;                                   // reuse buffer for p
    atomicAdd(&ssum[d * H + h], p);
}

// agg[dst, h, c] += feat[src, h, c] * p / ssum[dst, h]
__global__ void k_aggregate(const int* __restrict__ srcA, const int* __restrict__ dstA,
                            const float* __restrict__ p, const float* __restrict__ ssum,
                            const float* __restrict__ feat, float* __restrict__ agg,
                            int H, int C, int E, int EN) {
    int tid = blockIdx.x * blockDim.x + threadIdx.x;
    if (tid >= EN * C) return;
    int c = tid % C, e = tid / C;
    int s, d;
    if (e < E) { s = srcA[e]; d = dstA[e]; } else { s = d = e - E; }
    int F = H * C;
    for (int h = 0; h < H; ++h) {
        float alpha = p[e * H + h] / ssum[d * H + h];
        atomicAdd(&agg[(size_t)d * F + h * C + c],
                  feat[(size_t)s * F + h * C + c] * alpha);
    }
}

// layer-1 epilogue: h2 = elu(agg1 + b1)   (in place)
__global__ void k_bias_elu(float* __restrict__ a, const float* __restrict__ b1) {
    int tid = blockIdx.x * blockDim.x + threadIdx.x;
    if (tid >= N_NODES * HC1) return;
    float v = a[tid] + b1[tid % HC1];
    a[tid] = (v > 0.f) ? v : (__expf(v) - 1.f);
}

// pad W2[128,40] -> bf16 [128,48]
__global__ void k_w2pad(const float* __restrict__ W2, unsigned short* __restrict__ W2bf) {
    int tid = blockIdx.x * blockDim.x + threadIdx.x;
    if (tid >= 128 * OUT_PAD) return;
    int k = tid / OUT_PAD, col = tid % OUT_PAD;
    W2bf[tid] = f2bf(col < OUT_C ? W2[k * OUT_C + col] : 0.f);
}

// out = log_softmax(agg2 + b2) per node
__global__ void k_final(const float* __restrict__ agg2, const float* __restrict__ b2,
                        float* __restrict__ out) {
    int n = blockIdx.x * blockDim.x + threadIdx.x;
    if (n >= N_NODES) return;
    float v[OUT_C];
    float m = -1e30f;
    for (int c = 0; c < OUT_C; ++c) {
        v[c] = agg2[(size_t)n * OUT_C + c] + b2[c];
        m = fmaxf(m, v[c]);
    }
    float s = 0.f;
    for (int c = 0; c < OUT_C; ++c) s += __expf(v[c] - m);
    float ls = __logf(s);
    for (int c = 0; c < OUT_C; ++c) out[(size_t)n * OUT_C + c] = v[c] - m - ls;
}

// ============================================================================
extern "C" void kernel_launch(void* const* d_in, const int* in_sizes, int n_in,
                              void* d_out, int out_size, void* d_ws, size_t ws_size,
                              hipStream_t stream) {
    const float* x   = (const float*)d_in[0];
    const int*   ei  = (const int*)d_in[1];
    const float* W1  = (const float*)d_in[2];
    const float* aS1 = (const float*)d_in[3];
    const float* aD1 = (const float*)d_in[4];
    const float* b1  = (const float*)d_in[5];
    const float* W2  = (const float*)d_in[6];
    const float* aS2 = (const float*)d_in[7];
    const float* aD2 = (const float*)d_in[8];
    const float* b2  = (const float*)d_in[9];
    float* out = (float*)d_out;

    const int E  = in_sizes[1] / 2;
    const int EN = E + N_NODES;
    const int* esrc = ei;
    const int* edst = ei + E;

    // ---- workspace carve-up (all region sizes multiple of 4 floats) ----
    float* ws = (float*)d_ws;
    size_t o = 0;
    float*        h1    = ws + o; o += (size_t)N_NODES * HC1;
    float*        alS1v = ws + o; o += (size_t)N_NODES * H1;
    float*        alD1v = ws + o; o += (size_t)N_NODES * H1;
    unsigned int* mE1   = (unsigned int*)(ws + o); o += (size_t)N_NODES * H1;
    float*        ssum1 = ws + o; o += (size_t)N_NODES * H1;
    float*        elog1 = ws + o; o += (size_t)EN * H1;
    float*        agg1  = ws + o; o += (size_t)N_NODES * HC1;   // becomes h2 after bias+elu
    float*        g     = ws + o; o += (size_t)N_NODES * OUT_C;
    float*        alS2v = ws + o; o += (size_t)N_NODES;
    float*        alD2v = ws + o; o += (size_t)N_NODES;
    unsigned int* mE2   = (unsigned int*)(ws + o); o += (size_t)N_NODES;
    float*        ssum2 = ws + o; o += (size_t)N_NODES;
    float*        elog2 = ws + o; o += (size_t)EN;
    float*        agg2  = ws + o; o += (size_t)N_NODES * OUT_C;
    unsigned short* W2bf = (unsigned short*)(ws + o); // 128*48 ushorts

    auto cdiv = [](long a, long b) { return (int)((a + b - 1) / b); };
    const int B = 256;
    const unsigned int NEG_INF_ENC = 0x007FFFFFu;   // encF(-inf)

    // ---- per-call init (graph-capturable) ----
    k_fill_u32<<<cdiv(N_NODES * H1, B), B, 0, stream>>>(mE1, N_NODES * H1, NEG_INF_ENC);
    k_fill_u32<<<cdiv(N_NODES, B), B, 0, stream>>>(mE2, N_NODES, NEG_INF_ENC);
    hipMemsetAsync(ssum1, 0, (size_t)N_NODES * H1 * 4, stream);
    hipMemsetAsync(agg1,  0, (size_t)N_NODES * HC1 * 4, stream);
    hipMemsetAsync(ssum2, 0, (size_t)N_NODES * 4, stream);
    hipMemsetAsync(agg2,  0, (size_t)N_NODES * OUT_C * 4, stream);
    k_w2pad<<<cdiv(128 * OUT_PAD, B), B, 0, stream>>>(W2, W2bf);

    // ---- layer 1 ----
    gemm1_wmma<<<cdiv(N_NODES, 128), B, 0, stream>>>(x, W1, h1);
    k_al<<<cdiv((long)N_NODES * H1, B), B, 0, stream>>>(h1, aS1, aD1, alS1v, alD1v, H1, C1);
    k_logit_max<<<cdiv((long)EN * H1, B), B, 0, stream>>>(esrc, edst, alS1v, alD1v, elog1, mE1, H1, E, EN);
    k_expsum<<<cdiv((long)EN * H1, B), B, 0, stream>>>(esrc, edst, elog1, mE1, ssum1, H1, E, EN);
    k_aggregate<<<cdiv((long)EN * C1, B), B, 0, stream>>>(esrc, edst, elog1, ssum1, h1, agg1, H1, C1, E, EN);
    k_bias_elu<<<cdiv((long)N_NODES * HC1, B), B, 0, stream>>>(agg1, b1);

    // ---- layer 2 ----
    gemm2_wmma<<<cdiv(N_NODES, 128), B, 0, stream>>>(agg1, W2bf, g);
    k_al<<<cdiv(N_NODES, B), B, 0, stream>>>(g, aS2, aD2, alS2v, alD2v, 1, OUT_C);
    k_logit_max<<<cdiv(EN, B), B, 0, stream>>>(esrc, edst, alS2v, alD2v, elog2, mE2, 1, E, EN);
    k_expsum<<<cdiv(EN, B), B, 0, stream>>>(esrc, edst, elog2, mE2, ssum2, 1, E, EN);
    k_aggregate<<<cdiv((long)EN * OUT_C, B), B, 0, stream>>>(esrc, edst, elog2, ssum2, g, agg2, 1, OUT_C, E, EN);
    k_final<<<cdiv(N_NODES, B), B, 0, stream>>>(agg2, b2, out);
}
